// RESULinear_10977936408699
// MI455X (gfx1250) — compile-verified, matrix operand order checked by reading, not run on hardware
//
#include <hip/hip_runtime.h>

// ---------------------------------------------------------------------------
// RESULinear for MI455X (gfx1250, wave32, WMMA + Tensor Data Mover)
//
// Y[m,n] = sum_k X[m,k] * W_eff[n,k] + bias[n]
//   W_eff = mask ? (wq - qzero)*qscale : theta
// M = 8192, N = 4096, K = 4096 -> 275 GFLOP GEMM, compute-bound
// (HBM floor ~14us at 23.3 TB/s). Matrix math on v_wmma_f32_16x16x32_bf16
// with a 3-term hi/lo bf16 split (Ahi*Bhi + Ahi*Blo + Alo*Bhi, fp32 accum)
// for near-fp32 accuracy. W tiles are staged LDS-side by the TDM
// (tensor_load_to_lds), double-buffered and synced with s_wait_tensorcnt.
// ---------------------------------------------------------------------------

#define IN_F   4096
#define OUT_F  4096
#define BATCH  4
#define SEQ    2048
#define MROWS  (BATCH * SEQ)   // 8192

#define BM 128
#define BN 128
#define BK 32
#define LDSPITCH 40   // 32 bf16 + 16B pad = 80B rows -> conflict-free fragments

typedef __attribute__((ext_vector_type(16))) __bf16 v16bf;
typedef __attribute__((ext_vector_type(8)))  __bf16 v8bf;
typedef __attribute__((ext_vector_type(8)))  float  v8f;
typedef __attribute__((ext_vector_type(4)))  float  v4f;
typedef __attribute__((ext_vector_type(4)))  unsigned int u32x4;
typedef __attribute__((ext_vector_type(8)))  unsigned int u32x8;

// ---------------------------------------------------------------------------
// Prep: W_eff -> bf16 hi/lo planes in workspace (row-major [OUT_F][IN_F]).
// ---------------------------------------------------------------------------
__global__ __launch_bounds__(256) void resu_prep_weights(
    const int* __restrict__ wq, const float* __restrict__ qscale,
    const float* __restrict__ qzero, const float* __restrict__ theta,
    const unsigned char* __restrict__ mask,
    __bf16* __restrict__ whi, __bf16* __restrict__ wlo)
{
    long long base = ((long long)blockIdx.x * 256 + threadIdx.x) * 4;
    int o = (int)(base >> 12);           // base / IN_F (4 elems share a row)
    float s = qscale[o];
    float z = qzero[o];
#pragma unroll
    for (int j = 0; j < 4; ++j) {
        long long e = base + j;
        float w = mask[e] ? ((float)wq[e] - z) * s : theta[e];
        __bf16 h = (__bf16)w;
        whi[e] = h;
        wlo[e] = (__bf16)(w - (float)h);
    }
}

// ---------------------------------------------------------------------------
// TDM: DMA one 128-row x 32-col bf16 tile (row stride IN_F) into LDS.
// D# per CDNA5 ISA ch.8; pad_enable inserts 16B after every 64B row so the
// LDS image lands with an 80B pitch == LDSPITCH, conflict-free for the
// per-lane WMMA fragment gathers. 2D tensor -> groups 2/3 omitted (NULL).
// ---------------------------------------------------------------------------
__device__ __forceinline__ unsigned lds_off(const void* p) {
    // generic pointer to LDS: low 32 bits are the LDS byte offset
    return (unsigned)(unsigned long long)(uintptr_t)p;
}

__device__ __forceinline__ void tdm_load_w_tile(const __bf16* tile_start,
                                                unsigned lds_addr)
{
    unsigned long long ga = (unsigned long long)(uintptr_t)tile_start;
    u32x4 g0;
    g0[0] = 1u;                                        // count=1, user mode
    g0[1] = lds_addr;                                  // lds_addr
    g0[2] = (unsigned)(ga & 0xFFFFFFFFu);              // global_addr[31:0]
    g0[3] = (unsigned)((ga >> 32) & 0x1FFFFFFu)        // global_addr[56:32]
          | (2u << 30);                                // type = 2 (image)
    u32x8 g1;
    g1[0] = (1u << 16)        // data_size = 2 bytes
          | (1u << 20)        // pad_enable
          | (3u << 22)        // pad_interval: 16 DWORDs (64B = one tile row)
          | (3u << 25);       // pad_amount:   4 DWORDs (16B)
    g1[1] = (IN_F & 0xFFFFu) << 16;                    // tensor_dim0[15:0]
    g1[2] = ((unsigned)IN_F >> 16)                     // tensor_dim0[31:16]
          | ((OUT_F & 0xFFFFu) << 16);                 // tensor_dim1[15:0]
    g1[3] = ((unsigned)OUT_F >> 16)                    // tensor_dim1[31:16]
          | (32u << 16);                               // tile_dim0 = 32 elems
    g1[4] = 128u;                                      // tile_dim1 = 128 rows
    g1[5] = (unsigned)IN_F;                            // dim0_stride[31:0]
    g1[6] = 0u;                                        // dim0_stride hi, dim1_stride lo
    g1[7] = 0u;
    asm volatile("tensor_load_to_lds %0, %1" :: "s"(g0), "s"(g1) : "memory");
}

// ---------------------------------------------------------------------------
// GEMM: block 128x128, K-step 32, 8 waves; each wave computes 32x64
// (2x4 WMMA tiles, fp32 accumulators). 3 WMMAs per tile per K-step.
// Wave0/1 drive the TDM for the W hi/lo planes (double-buffered).
// ---------------------------------------------------------------------------
__global__ __launch_bounds__(256) void resu_gemm_wmma(
    const float* __restrict__ x,
    const __bf16* __restrict__ whi, const __bf16* __restrict__ wlo,
    const float* __restrict__ bias, float* __restrict__ out)
{
    __shared__ __align__(16) __bf16 sAhi[BM * LDSPITCH];
    __shared__ __align__(16) __bf16 sAlo[BM * LDSPITCH];
    __shared__ __align__(16) __bf16 sBhi[2][BN * LDSPITCH];
    __shared__ __align__(16) __bf16 sBlo[2][BN * LDSPITCH];

    const int tid  = threadIdx.x;
    const int lane = tid & 31;          // wave32 lane
    const int wave = tid >> 5;          // 0..7
    const int wm0  = (wave & 3) * 32;   // wave M origin in block tile
    const int wn0  = (wave >> 2) * 64;  // wave N origin in block tile
    const int r16  = lane & 15;         // fragment row (M or N)
    const int k0   = (lane >> 4) * 8;   // fragment K-chunk base (ISA layout)

    // cooperative X staging: thread -> (row, 16-wide K half)
    const int ldrow = tid >> 1;         // 0..127
    const int ldcol = (tid & 1) * 16;   // 0 or 16

    const long long gm = (long long)blockIdx.y * BM;   // output row base
    const int       gn = blockIdx.x * BN;              // output col base

    const float*  xrow     = x   + (gm + ldrow) * IN_F + ldcol;
    const __bf16* wtile_hi = whi + (long long)gn * IN_F;   // row gn, col 0
    const __bf16* wtile_lo = wlo + (long long)gn * IN_F;

    v8f acc[2][4];
#pragma unroll
    for (int mt = 0; mt < 2; ++mt)
#pragma unroll
        for (int nt = 0; nt < 4; ++nt)
            acc[mt][nt] = (v8f){0.f, 0.f, 0.f, 0.f, 0.f, 0.f, 0.f, 0.f};

    // prologue: DMA the first W tiles into buffer 0
    if (wave == 0)      tdm_load_w_tile(wtile_hi, lds_off(&sBhi[0][0]));
    else if (wave == 1) tdm_load_w_tile(wtile_lo, lds_off(&sBlo[0][0]));

    int buf = 0;
    for (int kt = 0; kt < IN_F; kt += BK) {
        // ---- stage X (fp32 -> bf16 hi/lo) into LDS ----
        v4f x0 = *(const v4f*)(xrow + kt);
        v4f x1 = *(const v4f*)(xrow + kt + 4);
        v4f x2 = *(const v4f*)(xrow + kt + 8);
        v4f x3 = *(const v4f*)(xrow + kt + 12);
        if (kt + BK < IN_F)
            __builtin_prefetch(xrow + kt + BK, 0, 0);

        float xv[16];
        *((v4f*)xv + 0) = x0; *((v4f*)xv + 1) = x1;
        *((v4f*)xv + 2) = x2; *((v4f*)xv + 3) = x3;
        v8bf ah[2], al[2];
#pragma unroll
        for (int c = 0; c < 2; ++c)
#pragma unroll
            for (int i = 0; i < 8; ++i) {
                float v  = xv[c * 8 + i];
                __bf16 h = (__bf16)v;
                ah[c][i] = h;
                al[c][i] = (__bf16)(v - (float)h);
            }
        __bf16* dA = &sAhi[ldrow * LDSPITCH + ldcol];
        *(v8bf*)(dA)     = ah[0];
        *(v8bf*)(dA + 8) = ah[1];
        __bf16* dAl = &sAlo[ldrow * LDSPITCH + ldcol];
        *(v8bf*)(dAl)     = al[0];
        *(v8bf*)(dAl + 8) = al[1];

        // W tiles for this step have landed (per-wave TENSORcnt; barrier
        // below publishes them to the whole workgroup)
        __builtin_amdgcn_s_wait_tensorcnt(0);
        __syncthreads();

        // kick off the DMA for the next K-step into the other buffer,
        // overlapping with this step's WMMA work
        if (kt + BK < IN_F) {
            if (wave == 0)
                tdm_load_w_tile(wtile_hi + (kt + BK), lds_off(&sBhi[buf ^ 1][0]));
            else if (wave == 1)
                tdm_load_w_tile(wtile_lo + (kt + BK), lds_off(&sBlo[buf ^ 1][0]));
        }

        // ---- gather fragments (per-lane: row r16, chunks k0 and k0+16) ----
        v16bf aH[2], aL[2], bH[4], bL[4];
#pragma unroll
        for (int mt = 0; mt < 2; ++mt) {
            const __bf16* rh = &sAhi[(wm0 + mt * 16 + r16) * LDSPITCH];
            const __bf16* rl = &sAlo[(wm0 + mt * 16 + r16) * LDSPITCH];
            v8bf c0 = *(const v8bf*)(rh + k0);
            v8bf c1 = *(const v8bf*)(rh + k0 + 16);
            aH[mt] = __builtin_shufflevector(c0, c1, 0,1,2,3,4,5,6,7,8,9,10,11,12,13,14,15);
            c0 = *(const v8bf*)(rl + k0);
            c1 = *(const v8bf*)(rl + k0 + 16);
            aL[mt] = __builtin_shufflevector(c0, c1, 0,1,2,3,4,5,6,7,8,9,10,11,12,13,14,15);
        }
#pragma unroll
        for (int nt = 0; nt < 4; ++nt) {
            const __bf16* rh = &sBhi[buf][(wn0 + nt * 16 + r16) * LDSPITCH];
            const __bf16* rl = &sBlo[buf][(wn0 + nt * 16 + r16) * LDSPITCH];
            v8bf c0 = *(const v8bf*)(rh + k0);
            v8bf c1 = *(const v8bf*)(rh + k0 + 16);
            bH[nt] = __builtin_shufflevector(c0, c1, 0,1,2,3,4,5,6,7,8,9,10,11,12,13,14,15);
            c0 = *(const v8bf*)(rl + k0);
            c1 = *(const v8bf*)(rl + k0 + 16);
            bL[nt] = __builtin_shufflevector(c0, c1, 0,1,2,3,4,5,6,7,8,9,10,11,12,13,14,15);
        }

        // ---- 2x4 tiles x 3 split terms = 24 WMMAs per K-step per wave ----
#pragma unroll
        for (int mt = 0; mt < 2; ++mt)
#pragma unroll
            for (int nt = 0; nt < 4; ++nt) {
                acc[mt][nt] = __builtin_amdgcn_wmma_f32_16x16x32_bf16(
                    false, aH[mt], false, bH[nt], (short)0, acc[mt][nt], false, false);
                acc[mt][nt] = __builtin_amdgcn_wmma_f32_16x16x32_bf16(
                    false, aH[mt], false, bL[nt], (short)0, acc[mt][nt], false, false);
                acc[mt][nt] = __builtin_amdgcn_wmma_f32_16x16x32_bf16(
                    false, aL[mt], false, bH[nt], (short)0, acc[mt][nt], false, false);
            }

        __syncthreads();   // compute done: sA free, sB[buf] free for the DMA after next
        buf ^= 1;
    }

    // ---- epilogue: D layout VGPR r -> row (lane<16 ? r : r+8), col lane&15 ----
    const int rhalf = (lane >> 4) * 8;
#pragma unroll
    for (int nt = 0; nt < 4; ++nt) {
        const int col = gn + wn0 + nt * 16 + r16;
        const float b = bias[col];
#pragma unroll
        for (int mt = 0; mt < 2; ++mt) {
            const long long rowb = gm + wm0 + mt * 16 + rhalf;
#pragma unroll
            for (int r = 0; r < 8; ++r)
                out[(rowb + r) * OUT_F + col] = acc[mt][nt][r] + b;
        }
    }
}

// ---------------------------------------------------------------------------
// Launch
// ---------------------------------------------------------------------------
extern "C" void kernel_launch(void* const* d_in, const int* in_sizes, int n_in,
                              void* d_out, int out_size, void* d_ws, size_t ws_size,
                              hipStream_t stream) {
    (void)in_sizes; (void)n_in; (void)out_size;

    const float*         x      = (const float*)d_in[0];
    const int*           wq     = (const int*)d_in[1];
    const float*         qscale = (const float*)d_in[2];
    const float*         qzero  = (const float*)d_in[3];
    const float*         theta  = (const float*)d_in[4];
    const unsigned char* mask   = (const unsigned char*)d_in[5]; // jnp.bool_: 1 byte
    const float*         bias   = (const float*)d_in[6];
    float*               out    = (float*)d_out;

    const size_t wElems = (size_t)OUT_F * IN_F;
    if (ws_size < wElems * 2 * sizeof(__bf16)) return;  // need 64 MB scratch
    __bf16* whi = (__bf16*)d_ws;
    __bf16* wlo = whi + wElems;

    // 1) dequant + mask-mix -> bf16 hi/lo weight planes
    resu_prep_weights<<<(int)(wElems / 4 / 256), 256, 0, stream>>>(
        wq, qscale, qzero, theta, mask, whi, wlo);

    // 2) WMMA GEMM with TDM-staged weights
    dim3 grid(OUT_F / BN, MROWS / BM);   // 32 x 64
    resu_gemm_wmma<<<grid, 256, 0, stream>>>(x, whi, wlo, bias, out);
}
